// TGN_31559419691805
// MI455X (gfx1250) — matrix operand order repaired and checked by reading, not compiled, not run on hardware
//
#include <hip/hip_runtime.h>
#include <cmath>

#define T_LEN     100000
#define DIN       128
#define DH        64
#define MAX_FRAME 20

typedef __attribute__((ext_vector_type(16))) __bf16 v16bf;
typedef __attribute__((ext_vector_type(8)))  float  v8f;

union FragU { v16bf v; uint4 q[2]; };

// ---------------------------------------------------------------------------
// Stage 1: fused  a_seq = sigmoid( [tanh(xA@WA+bA), tanh(xB@WB+bB)] @ Wa + ba )
// One wave per 16-row tile, 4 waves (128 threads) per block.
// W staged once per block into LDS, pre-swizzled into WMMA B-fragment layout.
// ---------------------------------------------------------------------------
__global__ __launch_bounds__(128)
void tgn_gemm_alpha_kernel(const float* __restrict__ xA,
                           const float* __restrict__ xB,
                           const float* __restrict__ WA,
                           const float* __restrict__ bA,
                           const float* __restrict__ WB,
                           const float* __restrict__ bB,
                           const float* __restrict__ Wa,
                           const float* __restrict__ ba,
                           float* __restrict__ a_seq)
{
    // fragW[net][kt][nt][lane] : 32 bytes (16 bf16) each = 32 KB total
    __shared__ uint4 fragW[2048];

    const int tid = threadIdx.x;

    // ---- cooperative fill: swizzle W (f32, row-major [K=128][N=64]) into
    //      bf16 B-fragments. Element j of lane's v16bf <-> K = kbase + j + (j >= 8 ? 8 : 0).
    for (int idx = tid; idx < 1024; idx += 128) {
        const int net = idx >> 9;            // 0 = WA, 1 = WB
        const int kt  = (idx >> 7) & 3;      // K-tile (32 K's each)
        const int nt  = (idx >> 5) & 3;      // N-tile (16 cols each)
        const int ln  = idx & 31;            // lane this fragment slice feeds
        const int ncol  = nt * 16 + (ln & 15);
        const int kbase = kt * 32 + 8 * (ln >> 4);
        const float* Wsrc = net ? WB : WA;
        FragU f;
#pragma unroll
        for (int j = 0; j < 16; ++j)
            f.v[j] = (__bf16)Wsrc[(kbase + j + (j >= 8 ? 8 : 0)) * DH + ncol];
        fragW[idx * 2]     = f.q[0];
        fragW[idx * 2 + 1] = f.q[1];
    }
    __syncthreads();

    const int lane = tid & 31;
    const int wave = tid >> 5;
    const int t0   = (blockIdx.x * 4 + wave) * 16;   // first row of this wave's tile
    if (t0 >= T_LEN) return;                          // T is a multiple of 16: full tiles only

    const int hi  = lane >> 4;       // half-wave select
    const int n   = lane & 15;       // column-within-tile / A-row
    const int row = t0 + n;          // A-matrix row this lane feeds

    // soft prefetch of the next tile's activations (global_prefetch_b8)
    if (row + 64 < T_LEN) {
        __builtin_prefetch(xA + (size_t)(row + 64) * DIN, 0, 1);
        __builtin_prefetch(xB + (size_t)(row + 64) * DIN, 0, 1);
    }

    // ---- load A fragments (16x32 bf16 per K-tile). Per lane: two contiguous
    //      8-float runs: K = kt*32+8*hi+[0..7] and K = kt*32+16+8*hi+[0..7].
    v16bf aA[4], aB[4];
#pragma unroll
    for (int kt = 0; kt < 4; ++kt) {
        const float* pA = xA + (size_t)row * DIN + kt * 32 + 8 * hi;
        const float* pB = xB + (size_t)row * DIN + kt * 32 + 8 * hi;
        float4 a0 = *(const float4*)(pA);
        float4 a1 = *(const float4*)(pA + 4);
        float4 a2 = *(const float4*)(pA + 16);
        float4 a3 = *(const float4*)(pA + 20);
        float4 b0 = *(const float4*)(pB);
        float4 b1 = *(const float4*)(pB + 4);
        float4 b2 = *(const float4*)(pB + 16);
        float4 b3 = *(const float4*)(pB + 20);
        FragU fa, fb;
        fa.v[0]=(__bf16)a0.x; fa.v[1]=(__bf16)a0.y; fa.v[2]=(__bf16)a0.z; fa.v[3]=(__bf16)a0.w;
        fa.v[4]=(__bf16)a1.x; fa.v[5]=(__bf16)a1.y; fa.v[6]=(__bf16)a1.z; fa.v[7]=(__bf16)a1.w;
        fa.v[8]=(__bf16)a2.x; fa.v[9]=(__bf16)a2.y; fa.v[10]=(__bf16)a2.z; fa.v[11]=(__bf16)a2.w;
        fa.v[12]=(__bf16)a3.x; fa.v[13]=(__bf16)a3.y; fa.v[14]=(__bf16)a3.z; fa.v[15]=(__bf16)a3.w;
        fb.v[0]=(__bf16)b0.x; fb.v[1]=(__bf16)b0.y; fb.v[2]=(__bf16)b0.z; fb.v[3]=(__bf16)b0.w;
        fb.v[4]=(__bf16)b1.x; fb.v[5]=(__bf16)b1.y; fb.v[6]=(__bf16)b1.z; fb.v[7]=(__bf16)b1.w;
        fb.v[8]=(__bf16)b2.x; fb.v[9]=(__bf16)b2.y; fb.v[10]=(__bf16)b2.z; fb.v[11]=(__bf16)b2.w;
        fb.v[12]=(__bf16)b3.x; fb.v[13]=(__bf16)b3.y; fb.v[14]=(__bf16)b3.z; fb.v[15]=(__bf16)b3.w;
        aA[kt] = fa.v;
        aB[kt] = fb.v;
    }

    // ---- accumulators pre-loaded with bias (bias depends only on N = column)
    v8f accA[4], accB[4];
#pragma unroll
    for (int nt = 0; nt < 4; ++nt) {
        const float bsA = bA[nt * 16 + n];
        const float bsB = bB[nt * 16 + n];
#pragma unroll
        for (int r = 0; r < 8; ++r) { accA[nt][r] = bsA; accB[nt][r] = bsB; }
    }

    // ---- 32 WMMAs: 2 nets x 4 N-tiles x K=128 (4 K-tiles of 32)
#pragma unroll
    for (int nt = 0; nt < 4; ++nt) {
#pragma unroll
        for (int kt = 0; kt < 4; ++kt) {
            FragU fw;
            int base = ((kt * 4 + nt) * 32 + lane) * 2;          // net 0
            fw.q[0] = fragW[base];
            fw.q[1] = fragW[base + 1];
            accA[nt] = __builtin_amdgcn_wmma_f32_16x16x32_bf16(
                false, aA[kt], false, fw.v, (short)0, accA[nt], false, false);
            base = ((16 + kt * 4 + nt) * 32 + lane) * 2;          // net 1
            fw.q[0] = fragW[base];
            fw.q[1] = fragW[base + 1];
            accB[nt] = __builtin_amdgcn_wmma_f32_16x16x32_bf16(
                false, aB[kt], false, fw.v, (short)0, accB[nt], false, false);
        }
    }

    // ---- tanh + Wa-dot. C/D layout: VGPR r at lane l -> (M = r + 8*hi, N = n).
    float s[8];
#pragma unroll
    for (int r = 0; r < 8; ++r) s[r] = 0.0f;
#pragma unroll
    for (int nt = 0; nt < 4; ++nt) {
        const float wa0 = Wa[nt * 16 + n];        // weight for hA columns
        const float wa1 = Wa[64 + nt * 16 + n];   // weight for hB columns
#pragma unroll
        for (int r = 0; r < 8; ++r)
            s[r] += tanhf(accA[nt][r]) * wa0 + tanhf(accB[nt][r]) * wa1;
    }
    // butterfly reduce over the 16 lanes of each half-wave (N dimension)
#pragma unroll
    for (int off = 1; off < 16; off <<= 1) {
#pragma unroll
        for (int r = 0; r < 8; ++r)
            s[r] += __shfl_xor(s[r], off, 32);
    }
    if (n == 0) {   // lane 0 -> rows 0..7, lane 16 -> rows 8..15
        const float bav = ba[0];
#pragma unroll
        for (int r = 0; r < 8; ++r) {
            float v = s[r] + bav;
            a_seq[t0 + 8 * hi + r] = 1.0f / (1.0f + expf(-v));
        }
    }
}

// ---------------------------------------------------------------------------
// Stage 2: serial TGN state machine (inherent 100k-step dependence chain).
// Single lane; faithful transcription of the reference scan.
// ---------------------------------------------------------------------------
__global__ void tgn_scan_kernel(const float* __restrict__ a_seq,
                                const int*   __restrict__ u,
                                const float* __restrict__ up,
                                const float* __restrict__ label,
                                float* __restrict__ y_out,
                                float* __restrict__ alpha_out,
                                float* __restrict__ uphat_out,
                                float* __restrict__ scal_out)
{
    if (threadIdx.x != 0 || blockIdx.x != 0) return;

    // Replicate _get_thres2 grid search (double to match host Python floats)
    double acc = 0.0, best_a = 0.001, min_score = 1e7;
    for (int it = 0; it < 100; ++it) {
        acc += 0.001;
        double y = 1.0 - pow(1.0 - acc, (double)MAX_FRAME);
        double sc = fabs(y - 0.6);
        if (min_score > sc) { min_score = sc; best_a = acc; }
    }
    float thres2[MAX_FRAME];
    for (int i = 0; i < MAX_FRAME; ++i)
        thres2[i] = (float)(1.0 - pow(1.0 - best_a, (double)(i + 1)));

    int   u_pre = 0;
    float up_pre = 0.0f, up_max = 0.0f, a_pre = 0.0f, y_pre = 0.0f;
    int   start_frame = 0, true_start = 0, label_frame = 0;
    float l_c = 0.0f, loss = 0.0f;
    int   cnt = 0;

    for (int i = 0; i < T_LEN; ++i) {
        const int   ui  = u[i];
        const float upi = up[i];
        const float ai  = a_seq[i];
        const float li  = label[i];

        if (u_pre == 0 && ui == 1) true_start = i;

        const bool rising = (up_pre < 0.5f) && (upi >= 0.5f);
        if (rising) { up_max = upi; start_frame = i; }
        up_max = fmaxf(up_max, upi);
        const float up_hat = (upi >= 0.5f) ? up_max : upi;

        const float alpha_ = up_hat * a_pre + (1.0f - up_hat) * ai;
        float y_ = alpha_ * up_hat + (1.0f - alpha_) * y_pre;
        if ((up_hat >= 0.5f) && (start_frame > 0) && (i - start_frame > MAX_FRAME))
            y_ = y_pre;

        const bool lab_hit = (li >= 0.5f);
        if (lab_hit) {
            l_c = (up_hat < 0.5f) ? -1.0f : (y_ - li) * (y_ - li);
            if (up_hat >= 0.5f) label_frame = i;
        }

        const bool fall    = (up_pre >= 0.5f) && (up_hat < 0.5f);
        const bool is_sent = (l_c == -1.0f);
        const bool is_nz   = (!is_sent) && (l_c != 0.0f);
        const bool is_e    = (!is_sent) && (l_c == 0.0f) && (y_pre >= 0.5f);
        const int  dur     = label_frame - true_start + 1;

        if (fall && is_nz && (dur >= 3) && (dur <= 40)) { loss += l_c; cnt += 1; }
        if (fall && is_e) {
            int tidx = i - start_frame;
            if (tidx < 0) tidx = 0;
            if (tidx > MAX_FRAME - 1) tidx = MAX_FRAME - 1;
            const float d = y_pre - thres2[tidx];
            loss += d * d;
            cnt += 1;
        }
        if (fall && (is_sent || is_nz || is_e)) l_c = 0.0f;
        if (fall) { start_frame = 0; y_ = 0.0f; }

        y_out[i]     = y_;
        alpha_out[i] = alpha_;
        uphat_out[i] = up_hat;

        u_pre = ui; up_pre = up_hat; a_pre = alpha_; y_pre = y_;
    }
    scal_out[0] = loss;
    scal_out[1] = (float)cnt;
}

// ---------------------------------------------------------------------------
extern "C" void kernel_launch(void* const* d_in, const int* in_sizes, int n_in,
                              void* d_out, int out_size, void* d_ws, size_t ws_size,
                              hipStream_t stream)
{
    const float* xA    = (const float*)d_in[0];
    const float* xB    = (const float*)d_in[1];
    const int*   u     = (const int*)  d_in[2];
    const float* up    = (const float*)d_in[3];
    const float* label = (const float*)d_in[4];
    const float* WA    = (const float*)d_in[5];
    const float* bA    = (const float*)d_in[6];
    const float* WB    = (const float*)d_in[7];
    const float* bB    = (const float*)d_in[8];
    const float* Wa    = (const float*)d_in[9];
    const float* ba    = (const float*)d_in[10];

    float* out   = (float*)d_out;            // [y(T) | alpha(T) | up_hat(T) | loss | cnt]
    float* a_seq = (float*)d_ws;             // T floats of scratch

    const int blocks = (T_LEN + 63) / 64;    // 4 waves/block, 16 rows/wave
    tgn_gemm_alpha_kernel<<<blocks, 128, 0, stream>>>(
        xA, xB, WA, bA, WB, bB, Wa, ba, a_seq);

    tgn_scan_kernel<<<1, 32, 0, stream>>>(
        a_seq, u, up, label,
        out, out + T_LEN, out + 2 * T_LEN, out + 3 * T_LEN);
}
